// SwiGLUExperts_8624294331062
// MI455X (gfx1250) — matrix operand order repaired and checked by reading, not compiled, not run on hardware
//
#include <hip/hip_runtime.h>
#include <hip/hip_bf16.h>

// ---------------------------------------------------------------------------
// MoE SwiGLU for MI455X (gfx1250): expert-grouped GEMMs on bf16 WMMA.
//   T=4096, A=2, E=8, D=1024, H=2816
//   Phase 1: bucket (t,a) pairs per expert
//   Phase 2: z = silu(X W_g^T) * (X W_u^T)  (bf16 out, fp32 accum via WMMA)
//   Phase 3: out = Z W_d^T                  (fp32 out)
// ---------------------------------------------------------------------------

typedef __attribute__((ext_vector_type(16))) __bf16 v16bf;
typedef __attribute__((ext_vector_type(8)))  __bf16 v8bf;
typedef __attribute__((ext_vector_type(8)))  float  v8f;
typedef __attribute__((ext_vector_type(4)))  float  f32x4;

#define T_TOK  4096
#define A_SLOT 2
#define NEXP   8
#define DDIM   1024
#define HDIM   2816
#define CAP    (T_TOK * A_SLOT)   // 8192 pairs (worst-case per expert)

#define BM 128
#define BN 64
#define BK 32
#define APITCH 40   // bf16 elems per LDS A row (padded; 80B, 16B aligned)
#define BPITCH 72   // bf16 elems per LDS Bt row (padded; 144B, 16B aligned)

__device__ __forceinline__ v8bf cvt8(f32x4 a, f32x4 b) {
  v8bf r;
  r[0] = (__bf16)a[0]; r[1] = (__bf16)a[1]; r[2] = (__bf16)a[2]; r[3] = (__bf16)a[3];
  r[4] = (__bf16)b[0]; r[5] = (__bf16)b[1]; r[6] = (__bf16)b[2]; r[7] = (__bf16)b[3];
  return r;
}

__device__ __forceinline__ v16bf cat16(v8bf a, v8bf b) {
  return __builtin_shufflevector(a, b, 0,1,2,3,4,5,6,7,8,9,10,11,12,13,14,15);
}

// Store one weight-column chunk (8 K-values for column n) transposed into LDS.
__device__ __forceinline__ void store_bcol(__bf16* B, int bk0, int bn, f32x4 a, f32x4 b) {
  B[(bk0 + 0) * BPITCH + bn] = (__bf16)a[0];
  B[(bk0 + 1) * BPITCH + bn] = (__bf16)a[1];
  B[(bk0 + 2) * BPITCH + bn] = (__bf16)a[2];
  B[(bk0 + 3) * BPITCH + bn] = (__bf16)a[3];
  B[(bk0 + 4) * BPITCH + bn] = (__bf16)b[0];
  B[(bk0 + 5) * BPITCH + bn] = (__bf16)b[1];
  B[(bk0 + 6) * BPITCH + bn] = (__bf16)b[2];
  B[(bk0 + 7) * BPITCH + bn] = (__bf16)b[3];
}

// ---------------------------------------------------------------------------
// Phase 1: routing
// ---------------------------------------------------------------------------
__global__ void k_zero_counts(int* __restrict__ counts) {
  if (threadIdx.x < NEXP) counts[threadIdx.x] = 0;
}

__global__ void k_assign(const int* __restrict__ expert_idx,
                         int* __restrict__ counts, int* __restrict__ lists) {
  int s = blockIdx.x * blockDim.x + threadIdx.x;
  if (s >= CAP) return;
  int e = expert_idx[s];
  int pos = atomicAdd(&counts[e], 1);
  lists[e * CAP + pos] = s;
}

// ---------------------------------------------------------------------------
// Phase 2: fused gate/up GEMM + silu*mul  ->  zbuf (bf16, [CAP][H])
// grid = (CAP/BM, H/BN, E), block = 256 (8 waves; 4x2 wave grid, 32x32/wave)
// ---------------------------------------------------------------------------
__global__ __launch_bounds__(256) void k_gateup(
    const float* __restrict__ x, const float* __restrict__ gate_w,
    const float* __restrict__ up_w, const int* __restrict__ counts,
    const int* __restrict__ lists, __bf16* __restrict__ zbuf)
{
  const int e = blockIdx.z;
  const int cnt = counts[e];
  const int mbase = blockIdx.x * BM;
  if (mbase >= cnt) return;
  const int nbase = blockIdx.y * BN;

  __shared__ __bf16 Alds[BM * APITCH];
  __shared__ __bf16 Bg[BK * BPITCH];
  __shared__ __bf16 Bu[BK * BPITCH];

  const int tid  = threadIdx.x;
  const int lane = tid & 31;
  const int wave = tid >> 5;
  const int wm   = wave >> 1;   // 0..3 -> m offset wm*32
  const int wn   = wave & 1;    // 0..1 -> n offset wn*32

  // A-load role: 2 threads per row, 16 fp32 each
  const int ar = tid >> 1;
  const int ac = (tid & 1) * 16;
  const int am_global = mbase + ar;
  const bool avalid = (am_global < cnt);
  const float* xrow = x;
  if (avalid) {
    int s = lists[e * CAP + am_global];
    xrow = x + (size_t)(s >> 1) * DDIM;   // token = s / A_SLOT
  }

  // B-load role: one column (H-row) per (tid&63), 8 K-values per thread
  const int bn  = tid & 63;
  const int bk0 = (tid >> 6) * 8;
  const float* grow = gate_w + ((size_t)e * HDIM + nbase + bn) * DDIM + bk0;
  const float* urow = up_w   + ((size_t)e * HDIM + nbase + bn) * DDIM + bk0;

  v8f accg[2][2] = {};
  v8f accu[2][2] = {};

  const int amrow = wm * 32 + (lane & 15);
  const int hh    = lane >> 4;            // K-half select for A fragment

  for (int kb = 0; kb < DDIM / BK; ++kb) {
    // ---- stage A tile (gathered token rows, fp32 -> bf16) ----
    {
      v8bf lo = {}, hi = {};
      if (avalid) {
        const float* p = xrow + kb * BK + ac;
        f32x4 f0 = *(const f32x4*)(p);
        f32x4 f1 = *(const f32x4*)(p + 4);
        f32x4 f2 = *(const f32x4*)(p + 8);
        f32x4 f3 = *(const f32x4*)(p + 12);
        lo = cvt8(f0, f1);
        hi = cvt8(f2, f3);
      }
      *(v8bf*)&Alds[ar * APITCH + ac]     = lo;
      *(v8bf*)&Alds[ar * APITCH + ac + 8] = hi;
    }
    // ---- stage B tiles transposed: Bt[k][n] ----
    {
      const float* p = grow + kb * BK;
      f32x4 g0 = *(const f32x4*)(p);
      f32x4 g1 = *(const f32x4*)(p + 4);
      p = urow + kb * BK;
      f32x4 u0 = *(const f32x4*)(p);
      f32x4 u1 = *(const f32x4*)(p + 4);
      store_bcol(Bg, bk0, bn, g0, g1);
      store_bcol(Bu, bk0, bn, u0, u1);
    }
    __syncthreads();

    // ---- fragments + WMMA ----
    v16bf afrag[2];
#pragma unroll
    for (int mi = 0; mi < 2; ++mi) {
      const __bf16* ap = &Alds[(amrow + mi * 16) * APITCH];
      v8bf a0 = *(const v8bf*)(ap + hh * 8);        // K = 8h .. 8h+7
      v8bf a1 = *(const v8bf*)(ap + 16 + hh * 8);   // K = 16+8h .. 16+8h+7
      afrag[mi] = cat16(a0, a1);
    }
#pragma unroll
    for (int ni = 0; ni < 2; ++ni) {
      const int nf = wn * 32 + ni * 16;
      v16bf bgf, buf;
#pragma unroll
      for (int s = 0; s < 16; ++s) {
        const int kk = (s < 8) ? (hh * 8 + s) : (16 + hh * 8 + (s - 8));
        bgf[s] = Bg[kk * BPITCH + nf + (lane & 15)];
        buf[s] = Bu[kk * BPITCH + nf + (lane & 15)];
      }
#pragma unroll
      for (int mi = 0; mi < 2; ++mi) {
        accg[mi][ni] = __builtin_amdgcn_wmma_f32_16x16x32_bf16(
            false, afrag[mi], false, bgf, (short)0, accg[mi][ni], false, false);
        accu[mi][ni] = __builtin_amdgcn_wmma_f32_16x16x32_bf16(
            false, afrag[mi], false, buf, (short)0, accu[mi][ni], false, false);
      }
    }
    __syncthreads();
  }

  // ---- epilogue: silu(gate)*up -> bf16 zbuf, scattered by pair index ----
#pragma unroll
  for (int mi = 0; mi < 2; ++mi) {
    const int mrow0 = mbase + wm * 32 + mi * 16 + hh * 8;
#pragma unroll
    for (int ni = 0; ni < 2; ++ni) {
      const int ncol = nbase + wn * 32 + ni * 16 + (lane & 15);
#pragma unroll
      for (int v = 0; v < 8; ++v) {
        int m = mrow0 + v;
        if (m < cnt) {
          int s = lists[e * CAP + m];
          float g = accg[mi][ni][v];
          float u = accu[mi][ni][v];
          float zv = (g / (1.0f + __expf(-g))) * u;
          zbuf[(size_t)s * HDIM + ncol] = (__bf16)zv;
        }
      }
    }
  }
}

// ---------------------------------------------------------------------------
// Phase 3: out = Z W_d^T   (A already bf16 in zbuf; output fp32)
// grid = (CAP/BM, D/BN, E), block = 256
// ---------------------------------------------------------------------------
__global__ __launch_bounds__(256) void k_down(
    const __bf16* __restrict__ zbuf, const float* __restrict__ down_w,
    const int* __restrict__ counts, const int* __restrict__ lists,
    float* __restrict__ out)
{
  const int e = blockIdx.z;
  const int cnt = counts[e];
  const int mbase = blockIdx.x * BM;
  if (mbase >= cnt) return;
  const int nbase = blockIdx.y * BN;

  __shared__ __bf16 Alds[BM * APITCH];
  __shared__ __bf16 Bl[BK * BPITCH];

  const int tid  = threadIdx.x;
  const int lane = tid & 31;
  const int wave = tid >> 5;
  const int wm   = wave >> 1;
  const int wn   = wave & 1;

  const int ar = tid >> 1;
  const int ac = (tid & 1) * 16;
  const int am_global = mbase + ar;
  const bool avalid = (am_global < cnt);
  const __bf16* zrow = zbuf;
  if (avalid) {
    int s = lists[e * CAP + am_global];
    zrow = zbuf + (size_t)s * HDIM;
  }

  const int bn  = tid & 63;
  const int bk0 = (tid >> 6) * 8;
  const float* drow = down_w + ((size_t)e * DDIM + nbase + bn) * HDIM + bk0;

  v8f acc[2][2] = {};

  const int amrow = wm * 32 + (lane & 15);
  const int hh    = lane >> 4;

  for (int kb = 0; kb < HDIM / BK; ++kb) {
    {
      v8bf lo = {}, hi = {};
      if (avalid) {
        const __bf16* p = zrow + kb * BK + ac;
        lo = *(const v8bf*)(p);
        hi = *(const v8bf*)(p + 8);
      }
      *(v8bf*)&Alds[ar * APITCH + ac]     = lo;
      *(v8bf*)&Alds[ar * APITCH + ac + 8] = hi;
    }
    {
      const float* p = drow + kb * BK;
      f32x4 d0 = *(const f32x4*)(p);
      f32x4 d1 = *(const f32x4*)(p + 4);
      store_bcol(Bl, bk0, bn, d0, d1);
    }
    __syncthreads();

    v16bf afrag[2];
#pragma unroll
    for (int mi = 0; mi < 2; ++mi) {
      const __bf16* ap = &Alds[(amrow + mi * 16) * APITCH];
      v8bf a0 = *(const v8bf*)(ap + hh * 8);
      v8bf a1 = *(const v8bf*)(ap + 16 + hh * 8);
      afrag[mi] = cat16(a0, a1);
    }
#pragma unroll
    for (int ni = 0; ni < 2; ++ni) {
      const int nf = wn * 32 + ni * 16;
      v16bf bf;
#pragma unroll
      for (int s = 0; s < 16; ++s) {
        const int kk = (s < 8) ? (hh * 8 + s) : (16 + hh * 8 + (s - 8));
        bf[s] = Bl[kk * BPITCH + nf + (lane & 15)];
      }
#pragma unroll
      for (int mi = 0; mi < 2; ++mi) {
        acc[mi][ni] = __builtin_amdgcn_wmma_f32_16x16x32_bf16(
            false, afrag[mi], false, bf, (short)0, acc[mi][ni], false, false);
      }
    }
    __syncthreads();
  }

#pragma unroll
  for (int mi = 0; mi < 2; ++mi) {
    const int mrow0 = mbase + wm * 32 + mi * 16 + hh * 8;
#pragma unroll
    for (int ni = 0; ni < 2; ++ni) {
      const int ncol = nbase + wn * 32 + ni * 16 + (lane & 15);
#pragma unroll
      for (int v = 0; v < 8; ++v) {
        int m = mrow0 + v;
        if (m < cnt) {
          int s = lists[e * CAP + m];
          out[(size_t)s * DDIM + ncol] = acc[mi][ni][v];
        }
      }
    }
  }
}

// ---------------------------------------------------------------------------
// Launch
// ---------------------------------------------------------------------------
extern "C" void kernel_launch(void* const* d_in, const int* in_sizes, int n_in,
                              void* d_out, int out_size, void* d_ws, size_t ws_size,
                              hipStream_t stream) {
  (void)in_sizes; (void)n_in; (void)out_size; (void)ws_size;
  const float* x      = (const float*)d_in[0];
  const int*   eidx   = (const int*)d_in[1];
  const float* up_w   = (const float*)d_in[2];
  const float* gate_w = (const float*)d_in[3];
  const float* down_w = (const float*)d_in[4];
  float* out = (float*)d_out;

  // workspace layout: [counts: 8 int][pad to 64B][lists: 8*8192 int][zbuf: 8192*2816 bf16]
  int* counts = (int*)d_ws;
  int* lists  = (int*)d_ws + 16;
  __bf16* zbuf = (__bf16*)((char*)d_ws + 64 + (size_t)NEXP * CAP * sizeof(int));

  k_zero_counts<<<1, 32, 0, stream>>>(counts);
  k_assign<<<CAP / 256, 256, 0, stream>>>(eidx, counts, lists);

  dim3 g2(CAP / BM, HDIM / BN, NEXP);   // (64, 44, 8) — inactive tiles early-exit
  k_gateup<<<g2, dim3(256), 0, stream>>>(x, gate_w, up_w, counts, lists, zbuf);

  dim3 g3(CAP / BM, DDIM / BN, NEXP);   // (64, 16, 8)
  k_down<<<g3, dim3(256), 0, stream>>>(zbuf, down_w, counts, lists, out);
}
